// Classifier_76209899700340
// MI455X (gfx1250) — matrix operand order repaired and checked by reading, not compiled, not run on hardware
//
#include <hip/hip_runtime.h>

typedef __attribute__((ext_vector_type(16))) _Float16 v16h;
typedef __attribute__((ext_vector_type(8)))  float    v8f;

#define N_GRAPHS 128
#define FDIM 128

// ---------------------------------------------------------------------------
// Utility: zero a float buffer
// ---------------------------------------------------------------------------
__global__ void gcn_zero(float* __restrict__ p, long long n) {
    long long t = (long long)blockIdx.x * blockDim.x + threadIdx.x;
    if (t < n) p[t] = 0.0f;
}

// ---------------------------------------------------------------------------
// Degree (in-degree over dst edges; self-loop +1 folded into dinv)
// ---------------------------------------------------------------------------
__global__ void gcn_degree(const int* __restrict__ dst, float* __restrict__ deg, int E) {
    int e = blockIdx.x * blockDim.x + threadIdx.x;
    if (e < E) atomicAdd(&deg[dst[e]], 1.0f);
}

__global__ void gcn_dinv(const float* __restrict__ deg, float* __restrict__ dinv, int n) {
    int i = blockIdx.x * blockDim.x + threadIdx.x;
    if (i < n) dinv[i] = rsqrtf(deg[i] + 1.0f);  // +1 = self loop; always > 0
}

// ---------------------------------------------------------------------------
// Weight transpose + f32->f16 convert:  Bt[n][k] = (f16) B[k][n]
// Tiny (<=16K elements), runs once per GEMM per launch; result is L2-resident.
// ---------------------------------------------------------------------------
__global__ void gcn_wtrans(const float* __restrict__ B, _Float16* __restrict__ Bt,
                           int K, int N) {
    int t = blockIdx.x * blockDim.x + threadIdx.x;
    if (t >= K * N) return;
    const int n = t / K, k = t - n * K;
    Bt[t] = (_Float16)B[k * N + n];
}

// ---------------------------------------------------------------------------
// WMMA row-strip GEMM:  C[M, NT*16] = f(A)[M,K] @ Bt^T   (+ cbias on C cols)
//   f(A) = relu(A + abias[k]) if ABIAS (fused activation of previous layer)
//   One wave computes a 16 x (NT*16) strip: A fragment loaded & converted once
//   per K-chunk, reused across NT column tiles (NT WMMAs per chunk).
//   Fragment layouts used here (wave32; contiguous k order, same in A and B):
//     A 16x32 f16: lane m=L&15, half=L>>4; elems 8g+j hold k = kk+16*half+8g+j
//     B 32x16 f16: lane n=L&15, half=L>>4; elems 0..15 hold k = kk+16*half+0..15
//       -> 16 contiguous f16 of transposed-weight row n: one 32B load
//     C 16x16 f32: vgpr v: m = 8*half+v, n = L&15
// ---------------------------------------------------------------------------
template<int NT, int K, bool ABIAS, bool CBIAS>
__global__ void gcn_gemm_wmma(const float* __restrict__ A, int lda,
                              const _Float16* __restrict__ Bt,   // [NT*16][K] f16
                              float* __restrict__ C, int ldc,
                              int M,
                              const float* __restrict__ abias,
                              const float* __restrict__ cbias) {
    const int lane = threadIdx.x & 31;
    const int wv   = threadIdx.x >> 5;
    const int wpb  = blockDim.x >> 5;
    const int tm   = blockIdx.x * wpb + wv;           // 16-row strip index
    if (tm >= (M >> 4)) return;  // wave-uniform scalar branch; EXEC stays all-1 for WMMA

    const int half = lane >> 4;
    const int l15  = lane & 15;
    const float* __restrict__ Arow = A + (long long)(tm * 16 + l15) * lda;

    v8f acc[NT];
#pragma unroll
    for (int t = 0; t < NT; ++t) acc[t] = (v8f){};

#pragma unroll
    for (int kk = 0; kk < K; kk += 32) {
        // ---- A fragment: two contiguous float4 pairs, convert once, reuse NT times
        v16h a;
#pragma unroll
        for (int g = 0; g < 2; ++g) {
            const int kbase = kk + (half << 4) + (g << 3);
            const float4 p0 = *(const float4*)(Arow + kbase);
            const float4 p1 = *(const float4*)(Arow + kbase + 4);
            float va[8] = {p0.x, p0.y, p0.z, p0.w, p1.x, p1.y, p1.z, p1.w};
#pragma unroll
            for (int j = 0; j < 8; ++j) {
                float f = va[j];
                if (ABIAS) f = fmaxf(f + abias[kbase + j], 0.0f);
                a[g * 8 + j] = (_Float16)f;
            }
        }
        // ---- NT column tiles: contiguous 32B f16 B fragment + WMMA each
#pragma unroll
        for (int t = 0; t < NT; ++t) {
            const int n = (t << 4) + l15;
            const v16h b = *(const v16h*)(Bt + (long long)n * K + kk + (half << 4));
            acc[t] = __builtin_amdgcn_wmma_f32_16x16x32_f16(
                false, a, false, b, (short)0, acc[t], false, false);
        }
    }

#pragma unroll
    for (int t = 0; t < NT; ++t) {
        const int n = (t << 4) + l15;
        const float cb = CBIAS ? cbias[n] : 0.0f;
#pragma unroll
        for (int v = 0; v < 8; ++v) {
            const int m = tm * 16 + (half << 3) + v;
            C[(long long)m * ldc + n] = acc[t][v] + cb;
        }
    }
}

// ---------------------------------------------------------------------------
// Edge scatter: out[dst] += h[src] * dinv[src]*dinv[dst].
// 32 threads per edge, float4 per thread (coalesced 16B gathers + f32 atomics)
// ---------------------------------------------------------------------------
__global__ void gcn_scatter(const int* __restrict__ src, const int* __restrict__ dst,
                            const float* __restrict__ dinv,
                            const float* __restrict__ h, float* __restrict__ out, int E) {
    const long long t = (long long)blockIdx.x * blockDim.x + threadIdx.x;
    const int e = (int)(t >> 5);
    if (e >= E) return;
    const int f = ((int)t & 31) << 2;
    const int s = src[e], d = dst[e];
    const float norm = dinv[s] * dinv[d];
    const float4 hv = *(const float4*)(h + (long long)s * FDIM + f);
    float* o = out + (long long)d * FDIM + f;
    atomicAdd(o + 0, hv.x * norm);
    atomicAdd(o + 1, hv.y * norm);
    atomicAdd(o + 2, hv.z * norm);
    atomicAdd(o + 3, hv.w * norm);
}

// Self-loop contribution: out[i,f] += h[i,f] * dinv[i]^2  (one writer per slot)
__global__ void gcn_selfloop(const float* __restrict__ h, const float* __restrict__ dinv,
                             float* __restrict__ out, long long n) {
    long long t = (long long)blockIdx.x * blockDim.x + threadIdx.x;
    if (t >= n) return;
    const float di = dinv[t >> 7];
    out[t] += h[t] * di * di;
}

// ---------------------------------------------------------------------------
// Global mean pool: pooled[g,f] += relu(agg[i,f] + b2[f]); cnt[g] += 1
// ---------------------------------------------------------------------------
__global__ void gcn_pool(const float* __restrict__ agg, const float* __restrict__ b2,
                         const int* __restrict__ batch,
                         float* __restrict__ pooled, float* __restrict__ cnt, int nodes) {
    long long t = (long long)blockIdx.x * blockDim.x + threadIdx.x;
    if (t >= (long long)nodes * FDIM) return;
    const int i = (int)(t >> 7), f = (int)t & 127;
    const int g = batch[i];
    const float v = fmaxf(agg[t] + b2[f], 0.0f);
    atomicAdd(&pooled[g * FDIM + f], v);
    if (f == 0) atomicAdd(&cnt[g], 1.0f);
}

__global__ void gcn_pool_finalize(float* __restrict__ pooled, const float* __restrict__ cnt) {
    int t = blockIdx.x * blockDim.x + threadIdx.x;  // N_GRAPHS*FDIM threads
    if (t >= N_GRAPHS * FDIM) return;
    pooled[t] /= fmaxf(cnt[t >> 7], 1.0f);
}

// ---------------------------------------------------------------------------
extern "C" void kernel_launch(void* const* d_in, const int* in_sizes, int n_in,
                              void* d_out, int out_size, void* d_ws, size_t ws_size,
                              hipStream_t stream) {
    const float* x     = (const float*)d_in[0];
    const int*   ei    = (const int*)d_in[1];   // [2, E] flat: row0=src, row1=dst
    const int*   batch = (const int*)d_in[2];
    const float* w1    = (const float*)d_in[3];
    const float* b1    = (const float*)d_in[4];
    const float* w2    = (const float*)d_in[5];
    const float* b2    = (const float*)d_in[6];
    const float* wc    = (const float*)d_in[7];
    const float* bc    = (const float*)d_in[8];
    float*       out   = (float*)d_out;

    const int E     = in_sizes[1] / 2;
    const int nodes = in_sizes[0] / FDIM;
    const int* esrc = ei;
    const int* edst = ei + E;

    // Workspace carve-up
    char* ws = (char*)d_ws;
    const long long NB = (long long)nodes * FDIM;           // elements per feature buffer
    float*     B1     = (float*)ws;      ws += NB * sizeof(float);
    float*     B2     = (float*)ws;      ws += NB * sizeof(float);
    float*     deg    = (float*)ws;      ws += (long long)nodes * sizeof(float);
    float*     dinv   = (float*)ws;      ws += (long long)nodes * sizeof(float);
    float*     pooled = (float*)ws;      ws += N_GRAPHS * FDIM * sizeof(float);
    float*     cnt    = (float*)ws;      ws += N_GRAPHS * sizeof(float);
    _Float16*  w1t    = (_Float16*)ws;   ws += FDIM * FDIM * sizeof(_Float16);
    _Float16*  w2t    = (_Float16*)ws;   ws += FDIM * FDIM * sizeof(_Float16);
    _Float16*  wct    = (_Float16*)ws;   // [16][128]

    const int T = 256;
    auto blks = [](long long n, int t) { return (unsigned)((n + t - 1) / t); };

    // 0) transpose+convert weights to f16 [N][K] (tiny; stays in L2)
    gcn_wtrans<<<blks(FDIM * FDIM, T), T, 0, stream>>>(w1, w1t, FDIM, FDIM);
    gcn_wtrans<<<blks(FDIM * FDIM, T), T, 0, stream>>>(w2, w2t, FDIM, FDIM);
    gcn_wtrans<<<blks(FDIM * 16, T), T, 0, stream>>>(wc, wct, FDIM, 16);

    // 1) degree -> dinv
    gcn_zero<<<blks(nodes, T), T, 0, stream>>>(deg, nodes);
    gcn_degree<<<blks(E, T), T, 0, stream>>>(edst, deg, E);
    gcn_dinv<<<blks(nodes, T), T, 0, stream>>>(deg, dinv, nodes);

    const long long strips = nodes / 16;   // 16-row strips per big GEMM

    // 2) h1 = x @ w1   (WMMA row-strip, 8 waves / block)
    gcn_gemm_wmma<8, FDIM, false, false><<<blks(strips, 8), T, 0, stream>>>(
        x, FDIM, w1t, B1, FDIM, nodes, nullptr, nullptr);

    // 3) agg1 = scatter(h1) + selfloop
    gcn_zero<<<blks(NB, T), T, 0, stream>>>(B2, NB);
    gcn_scatter<<<blks((long long)E * 32, T), T, 0, stream>>>(esrc, edst, dinv, B1, B2, E);
    gcn_selfloop<<<blks(NB, T), T, 0, stream>>>(B1, dinv, B2, NB);

    // 4) h2 = relu(agg1 + b1) @ w2  (bias+relu fused into A conversion)
    gcn_gemm_wmma<8, FDIM, true, false><<<blks(strips, 8), T, 0, stream>>>(
        B2, FDIM, w2t, B1, FDIM, nodes, b1, nullptr);

    // 5) agg2 = scatter(h2) + selfloop
    gcn_zero<<<blks(NB, T), T, 0, stream>>>(B2, NB);
    gcn_scatter<<<blks((long long)E * 32, T), T, 0, stream>>>(esrc, edst, dinv, B1, B2, E);
    gcn_selfloop<<<blks(NB, T), T, 0, stream>>>(B1, dinv, B2, NB);

    // 6) mean pool with relu(agg2 + b2)
    gcn_zero<<<blks(N_GRAPHS * FDIM + N_GRAPHS, T), T, 0, stream>>>(
        pooled, N_GRAPHS * FDIM + N_GRAPHS);
    gcn_pool<<<blks(NB, T), T, 0, stream>>>(B2, b2, batch, pooled, cnt, nodes);
    gcn_pool_finalize<<<blks(N_GRAPHS * FDIM, T), T, 0, stream>>>(pooled, cnt);

    // 7) out = pooled @ wc + bc   (M=128 -> 8 strips, NT=1, single block)
    gcn_gemm_wmma<1, FDIM, false, true><<<blks(N_GRAPHS / 16, 8), T, 0, stream>>>(
        pooled, FDIM, wct, out, 16, N_GRAPHS, nullptr, bc);
}
